// MultiHeadAttention_54400055771520
// MI455X (gfx1250) — matrix operand order repaired and checked by reading, not compiled, not run on hardware
//
#include <hip/hip_runtime.h>
#include <stdint.h>

#define B_    4
#define S_    1024
#define H_    1024
#define NH_   16
#define HD_   64
#define NREL  2047   /* 2*MAXPOS-1 */

typedef __bf16 bf16_t;
typedef __attribute__((ext_vector_type(16))) __bf16 v16bf;
typedef __attribute__((ext_vector_type(8)))  __bf16 v8bf;
typedef __attribute__((ext_vector_type(8)))  float  v8f;

__device__ __forceinline__ bf16_t f2bf(float f) {
  unsigned int u = __builtin_bit_cast(unsigned int, f);
  unsigned int r = (u + 0x7fffu + ((u >> 16) & 1u)) >> 16;
  return __builtin_bit_cast(bf16_t, (unsigned short)r);
}

__device__ __forceinline__ v8f vzero8() {
  v8f z;
#pragma unroll
  for (int i = 0; i < 8; ++i) z[i] = 0.0f;
  return z;
}

__device__ __forceinline__ v16bf pack16(v8bf lo, v8bf hi) {
  v16bf r;
#pragma unroll
  for (int i = 0; i < 8; ++i) { r[i] = lo[i]; r[8 + i] = hi[i]; }
  return r;
}

// A-fragment (16x32 bf16): lane m=lane&15, elems K = 16h+{0..15}
__device__ __forceinline__ v16bf frag_ld_a(const bf16_t* base, int ld, int row, int col) {
  const int lane = threadIdx.x & 31;
  const int m = lane & 15, h = lane >> 4;
  const bf16_t* p = base + (size_t)(row + m) * ld + col + 16 * h;
  v8bf lo = *(const v8bf*)(p);
  v8bf hi = *(const v8bf*)(p + 8);
  return pack16(lo, hi);
}

// B-fragment (32x16 bf16) from N-major [N][K] storage: lane n=lane&15, elems K = 16h+{0..15}
__device__ __forceinline__ v16bf frag_ld_b(const bf16_t* base, int ld, int row, int col) {
  const int lane = threadIdx.x & 31;
  const int n = lane & 15, h = lane >> 4;
  const bf16_t* p = base + (size_t)(row + n) * ld + col + 16 * h;
  v8bf lo = *(const v8bf*)(p);
  v8bf hi = *(const v8bf*)(p + 8);
  return pack16(lo, hi);
}

// B-fragment with per-lane row clamp (for dist_emb band edges)
__device__ __forceinline__ v16bf frag_ld_b_clamp(const bf16_t* base, int ld, int row, int col,
                                                 int rmax) {
  const int lane = threadIdx.x & 31;
  const int n = lane & 15, h = lane >> 4;
  int rr = row + n;
  rr = rr < 0 ? 0 : (rr > rmax ? rmax : rr);
  const bf16_t* p = base + (size_t)rr * ld + col + 16 * h;
  v8bf lo = *(const v8bf*)(p);
  v8bf hi = *(const v8bf*)(p + 8);
  return pack16(lo, hi);
}

__device__ __forceinline__ v8f wmma_bf16(v16bf a, v16bf b, v8f c) {
  return __builtin_amdgcn_wmma_f32_16x16x32_bf16(false, a, false, b, (short)0, c, false, false);
}

/* ---------------- kernel 0: dist_emb f32 -> bf16 ---------------- */
__global__ void cvt_emb(const float* __restrict__ e, bf16_t* __restrict__ out) {
  int i = blockIdx.x * 256 + threadIdx.x;
  if (i < NREL * HD_) out[i] = f2bf(e[i]);
}

/* ---------------- kernel 1: QKV projection GEMM ----------------
 * C[4096,1024] = X[4096,1024] @ W[1024,1024] + bias, output bf16.
 * vmode 0: out[bh][s][d]   (Q, K)
 * vmode 1: out[bh][d][s]   (V transposed, so attn B-fragments are row loads)
 * Block: 128 threads (4 waves), tile 64(M) x 64(N), K step 32.
 */
__global__ __launch_bounds__(128) void qkv_gemm(const float* __restrict__ X,
                                                const float* __restrict__ W,
                                                const float* __restrict__ bias,
                                                bf16_t* __restrict__ out, int vmode) {
  __shared__ bf16_t sX[64 * 32];  // [m][k]
  __shared__ bf16_t sW[64 * 32];  // [n][k] (transposed on store)
  const int tid = threadIdx.x;
  const int wave = tid >> 5, lane = tid & 31;
  const int m0 = blockIdx.x * 64;
  const int n0 = blockIdx.y * 64;

  v8f acc[4];
#pragma unroll
  for (int j = 0; j < 4; ++j) acc[j] = vzero8();

  for (int k0 = 0; k0 < H_; k0 += 32) {
    {
      int r = tid >> 1;
      int c = (tid & 1) * 16;
      const float* src = X + (size_t)(m0 + r) * H_ + k0 + c;
#pragma unroll
      for (int i = 0; i < 16; ++i) sX[r * 32 + c + i] = f2bf(src[i]);
      int kk = tid >> 2;
      int nb = (tid & 3) * 16;
      const float* wsrc = W + (size_t)(k0 + kk) * H_ + n0 + nb;
#pragma unroll
      for (int i = 0; i < 16; ++i) sW[(nb + i) * 32 + kk] = f2bf(wsrc[i]);
    }
    __syncthreads();
    v16bf a = frag_ld_a(sX, 32, wave * 16, 0);
#pragma unroll
    for (int j = 0; j < 4; ++j) {
      v16bf b = frag_ld_b(sW, 32, j * 16, 0);
      acc[j] = wmma_bf16(a, b, acc[j]);
    }
    __syncthreads();
  }

  const int n = lane & 15, h = lane >> 4;
#pragma unroll
  for (int j = 0; j < 4; ++j) {
    int ng = n0 + j * 16 + n;
    float bia = bias[ng];
    int head = ng >> 6, d = ng & 63;
#pragma unroll
    for (int v = 0; v < 8; ++v) {
      int row = m0 + wave * 16 + v + 8 * h;  // flat b*S+s
      int b = row >> 10, s = row & 1023;
      bf16_t val = f2bf(acc[j][v] + bia);
      size_t idx = vmode ? ((size_t)((b * NH_ + head) * HD_ + d) * S_ + s)
                         : ((size_t)((b * NH_ + head) * S_ + s) * HD_ + d);
      out[idx] = val;
    }
  }
}

/* ---------------- kernel 2: fused flash attention + relative bias ----------------
 * Grid: (S/64, B*NH). 128 threads = 4 waves; wave owns 16 query rows, BN=64 key block.
 */
__global__ __launch_bounds__(128) void attn(const bf16_t* __restrict__ Q,
                                            const bf16_t* __restrict__ K,
                                            const bf16_t* __restrict__ Vt,
                                            const bf16_t* __restrict__ E,
                                            const float* __restrict__ hyp,
                                            const float* __restrict__ mask,
                                            float* __restrict__ out) {
  __shared__ float  sD1[4][16][80];  // Q-strip x distance band (bias1)
  __shared__ float  sD2[4][64][32];  // K-block x distance band (bias2), per 16-key sub-block
  __shared__ bf16_t sP[4][16][64];   // probs for A-fragment reload

  const int tid = threadIdx.x, wave = tid >> 5, lane = tid & 31;
  const int n = lane & 15, h = lane >> 4;
  const int bh = blockIdx.y;
  const int bb = bh >> 4;
  const int head = bh & 15;
  const int l0 = blockIdx.x * 64 + wave * 16;

  const bf16_t* Qh = Q + (size_t)bh * S_ * HD_;
  const bf16_t* Kh = K + (size_t)bh * S_ * HD_;
  const bf16_t* Vh = Vt + (size_t)bh * HD_ * S_;
  const float* hypB = hyp + (size_t)bb * S_ * S_;
  const float* maskB = mask + (size_t)bb * S_;

  v16bf qf0 = frag_ld_a(Qh, HD_, l0, 0);
  v16bf qf1 = frag_ld_a(Qh, HD_, l0, 32);

  float mrow[8], lrow[8];
  v8f ctx[4];
#pragma unroll
  for (int v = 0; v < 8; ++v) { mrow[v] = -1e30f; lrow[v] = 0.0f; }
#pragma unroll
  for (int j = 0; j < 4; ++j) ctx[j] = vzero8();

  for (int r0 = 0; r0 < S_; r0 += 64) {
    if (r0 + 64 < S_) __builtin_prefetch(Kh + (size_t)(r0 + 64) * HD_, 0, 1);

    // ---- bias1 band GEMM: D1[a,t] = Q[l0+a,:] . E[cb+t,:],  t in [0,80)
    const int cb = l0 - r0 + 960;
#pragma unroll
    for (int t0 = 0; t0 < 5; ++t0) {
      v16bf e0 = frag_ld_b_clamp(E, HD_, cb + t0 * 16, 0, NREL - 1);
      v16bf e1 = frag_ld_b_clamp(E, HD_, cb + t0 * 16, 32, NREL - 1);
      v8f d1 = wmma_bf16(qf1, e1, wmma_bf16(qf0, e0, vzero8()));
#pragma unroll
      for (int v = 0; v < 8; ++v) sD1[wave][v + 8 * h][t0 * 16 + n] = d1[v];
    }

    // ---- raw scores + bias2 band GEMMs per 16-key sub-block
    v8f sacc[4];
#pragma unroll
    for (int j = 0; j < 4; ++j) {
      const int rcol = r0 + j * 16;
      v16bf kb0 = frag_ld_b(Kh, HD_, rcol, 0);   // B operand for q.k^T
      v16bf kb1 = frag_ld_b(Kh, HD_, rcol, 32);
      sacc[j] = wmma_bf16(qf1, kb1, wmma_bf16(qf0, kb0, vzero8()));

      v16bf ka0 = frag_ld_a(Kh, HD_, rcol, 0);   // A operand for bias2
      v16bf ka1 = frag_ld_a(Kh, HD_, rcol, 32);
      const int cb2 = l0 - rcol + 1008;
#pragma unroll
      for (int t0 = 0; t0 < 2; ++t0) {
        v16bf e0 = frag_ld_b_clamp(E, HD_, cb2 + t0 * 16, 0, NREL - 1);
        v16bf e1 = frag_ld_b_clamp(E, HD_, cb2 + t0 * 16, 32, NREL - 1);
        v8f d2 = wmma_bf16(ka1, e1, wmma_bf16(ka0, e0, vzero8()));
#pragma unroll
        for (int v = 0; v < 8; ++v) sD2[wave][j * 16 + v + 8 * h][t0 * 16 + n] = d2[v];
      }
    }

    // ---- assemble scores, online softmax
    v8f stile[4];
    v8f rmax;
#pragma unroll
    for (int v = 0; v < 8; ++v) rmax[v] = -1e30f;
#pragma unroll
    for (int j = 0; j < 4; ++j) {
      const int col = j * 16 + n;
      const int rg = r0 + col;
      const float mk = maskB[rg];
#pragma unroll
      for (int v = 0; v < 8; ++v) {
        const int a = v + 8 * h;
        float b1 = sD1[wave][a][a - col + 63];          // t in [0,78]
        float b2 = sD2[wave][col][a - n + 15];          // t in [0,30]
        float hv = hypB[(size_t)(l0 + a) * S_ + rg];
        float s = (sacc[j][v] + b1 + b2) * 0.125f + 0.5f * hv + mk;
        stile[j][v] = s;
        rmax[v] = fmaxf(rmax[v], s);
      }
    }
#pragma unroll
    for (int off = 1; off < 16; off <<= 1)
#pragma unroll
      for (int v = 0; v < 8; ++v) rmax[v] = fmaxf(rmax[v], __shfl_xor(rmax[v], off, 32));

    float scl[8];
#pragma unroll
    for (int v = 0; v < 8; ++v) {
      float mnew = fmaxf(mrow[v], rmax[v]);
      scl[v] = __expf(mrow[v] - mnew);
      mrow[v] = mnew;
    }
    v8f rsum = vzero8();
#pragma unroll
    for (int j = 0; j < 4; ++j)
#pragma unroll
      for (int v = 0; v < 8; ++v) {
        float p = __expf(stile[j][v] - mrow[v]);
        rsum[v] += p;
        sP[wave][v + 8 * h][j * 16 + n] = f2bf(p);
      }
#pragma unroll
    for (int off = 1; off < 16; off <<= 1)
#pragma unroll
      for (int v = 0; v < 8; ++v) rsum[v] += __shfl_xor(rsum[v], off, 32);
#pragma unroll
    for (int v = 0; v < 8; ++v) lrow[v] = lrow[v] * scl[v] + rsum[v];
#pragma unroll
    for (int j = 0; j < 4; ++j)
#pragma unroll
      for (int v = 0; v < 8; ++v) ctx[j][v] *= scl[v];

    // ---- probs @ V
    v16bf p0 = frag_ld_a(&sP[wave][0][0], 64, 0, 0);
    v16bf p1 = frag_ld_a(&sP[wave][0][0], 64, 0, 32);
#pragma unroll
    for (int j = 0; j < 4; ++j) {  // head-dim tile d0 = j*16
      v16bf v0 = frag_ld_b(Vh, S_, j * 16, r0);
      v16bf v1 = frag_ld_b(Vh, S_, j * 16, r0 + 32);
      ctx[j] = wmma_bf16(p1, v1, wmma_bf16(p0, v0, ctx[j]));
    }
  }

  // ---- epilogue: normalize and write [B,S,H]
#pragma unroll
  for (int j = 0; j < 4; ++j) {
    const int dg = head * HD_ + j * 16 + n;
#pragma unroll
    for (int v = 0; v < 8; ++v) {
      const int row = l0 + v + 8 * h;
      out[(size_t)(bb * S_ + row) * H_ + dg] = ctx[j][v] / lrow[v];
    }
  }
}

extern "C" void kernel_launch(void* const* d_in, const int* in_sizes, int n_in,
                              void* d_out, int out_size, void* d_ws, size_t ws_size,
                              hipStream_t stream) {
  const float* hidden = (const float*)d_in[0];
  const float* amask  = (const float*)d_in[1];
  const float* hyp    = (const float*)d_in[2];
  const float* Wq     = (const float*)d_in[3];
  const float* bq     = (const float*)d_in[4];
  const float* Wk     = (const float*)d_in[5];
  const float* bk     = (const float*)d_in[6];
  const float* Wv     = (const float*)d_in[7];
  const float* bv     = (const float*)d_in[8];
  const float* de     = (const float*)d_in[9];
  float* out = (float*)d_out;

  bf16_t* ws = (bf16_t*)d_ws;
  const size_t HSZ = (size_t)B_ * NH_ * S_ * HD_;  // 4 Mi elems
  bf16_t* qb = ws;
  bf16_t* kb = ws + HSZ;
  bf16_t* vb = ws + 2 * HSZ;
  bf16_t* eb = ws + 3 * HSZ;

  cvt_emb<<<(NREL * HD_ + 255) / 256, 256, 0, stream>>>(de, eb);

  dim3 g1(B_ * S_ / 64, H_ / 64);
  qkv_gemm<<<g1, 128, 0, stream>>>(hidden, Wq, bq, qb, 0);
  qkv_gemm<<<g1, 128, 0, stream>>>(hidden, Wk, bk, kb, 0);
  qkv_gemm<<<g1, 128, 0, stream>>>(hidden, Wv, bv, vb, 1);

  dim3 g2(S_ / 64, B_ * NH_);
  attn<<<g2, 128, 0, stream>>>(qb, kb, vb, eb, hyp, amask, out);
}
